// DualCrossAttention_concat_10754598109601
// MI455X (gfx1250) — matrix operand order repaired and checked by reading, not compiled, not run on hardware
//
#include <hip/hip_runtime.h>

// ---------------------------------------------------------------------------
// DualCrossAttention on MI455X (gfx1250, wave32, WMMA)
//
// Roofline: 302 MB mandatory f32 output (probs dominates) => ~13us HBM floor
// at 23.3 TB/s. ctx GEMMs are 34.4 GFLOP -> run on v_wmma_f32_16x16x32_f16
// (f16 in, f32 accumulate) so compute hides under the probs writeback.
// probs/ctx are streaming (written once, never re-read) -> nontemporal stores
// keep the 192MB L2 free for the 16MB of f16 key matrices (128x reuse).
// Unnormalized exp rows live in LDS (16 x 2048 f32 = 128KB, fits in 320KB/WGP)
// so exp() is evaluated exactly once per probs element.
// ---------------------------------------------------------------------------

typedef _Float16 h16 __attribute__((ext_vector_type(16)));
typedef _Float16 h8  __attribute__((ext_vector_type(8)));
typedef float    f8  __attribute__((ext_vector_type(8)));
typedef float    f4  __attribute__((ext_vector_type(4)));

#define ALPHA 0.01f
constexpr int BB = 8;     // batch
constexpr int NN = 2048;  // sequence length (both sides)
constexpr int DD = 256;   // feature dim
constexpr int PE_PITCH = NN + 4;  // LDS row pitch (floats) to dodge bank conflicts

// --------------------------------------------------------------------------
// K0: fold attention vectors: vecs[0]=W1@a1[:256], [1]=W1@a1[256:],
//                             [2]=W2@a2[:256], [3]=W2@a2[256:]
// (Wh_q is only ever dotted with a -> the query-side projection collapses to
//  a mat-vec and is never materialized.)
// --------------------------------------------------------------------------
__global__ void fold_kernel(const float* __restrict__ W1, const float* __restrict__ a1,
                            const float* __restrict__ W2, const float* __restrict__ a2,
                            float* __restrict__ vecs) {
  int which = blockIdx.x;          // 0..3
  int i = threadIdx.x;             // 0..255 : row of W
  const float* W = (which < 2) ? W1 : W2;
  const float* a = (which < 2) ? a1 : a2;
  const float* ah = a + ((which & 1) ? DD : 0);
  float s = 0.f;
  #pragma unroll 8
  for (int o = 0; o < DD; ++o) s += W[i * DD + o] * ah[o];
  vecs[which * DD + i] = s;
}

// --------------------------------------------------------------------------
// K1: score vectors. One wave32 per output row (length-256 dot).
//   tgt0: sq1 = X1 . vecs[0]   tgt1: sk1 = X2 . vecs[1]
//   tgt2: sq2 = X2 . vecs[2]   tgt3: sk2 = X1 . vecs[3]
// --------------------------------------------------------------------------
__global__ __launch_bounds__(256)
void score_kernel(const float* __restrict__ X1, const float* __restrict__ X2,
                  const float* __restrict__ vecs,
                  float* __restrict__ sq1, float* __restrict__ sk1,
                  float* __restrict__ sq2, float* __restrict__ sk2) {
  int w = blockIdx.x * 8 + (threadIdx.x >> 5);
  int lane = threadIdx.x & 31;
  int tgt = w >> 14;          // / 16384
  int row = w & 16383;
  const float* X = (tgt == 0 || tgt == 3) ? X1 : X2;
  const float* v = vecs + tgt * DD;
  const float* xr = X + (size_t)row * DD;
  float s = 0.f;
  #pragma unroll
  for (int t = 0; t < 8; ++t) {
    int k = lane + 32 * t;
    s += xr[k] * v[k];
  }
  #pragma unroll
  for (int off = 16; off > 0; off >>= 1) s += __shfl_xor(s, off, 32);
  if (lane == 0) {
    float* out = (tgt == 0) ? sq1 : (tgt == 1) ? sk1 : (tgt == 2) ? sq2 : sk2;
    out[row] = s;
  }
}

// --------------------------------------------------------------------------
// K2: key projections via WMMA:  whk1 = X2 @ W1,  whk2 = X1 @ W2
// Output stored TRANSPOSED f16: whk[b][feat][key], pitch NN, so the attention
// kernel's B-fragments are two contiguous 16B loads per lane.
// W is staged transposed in LDS (f16, pitch 264 halves) ONCE per block and
// amortized over M_ITERS m-tiles per wave (staging traffic 8x lower than one
// tile per block). Wave tile: 16(M) x 64(N); K loop 256/32 = 8 x 4 WMMAs.
// --------------------------------------------------------------------------
constexpr int M_ITERS = 8;   // m-tiles per wave; grid.x = 16384/16/2/M_ITERS = 64

__global__ __launch_bounds__(256)
void proj_kernel(const float* __restrict__ X1, const float* __restrict__ X2,
                 const float* __restrict__ W1, const float* __restrict__ W2,
                 _Float16* __restrict__ whk1, _Float16* __restrict__ whk2) {
  extern __shared__ char smem[];
  _Float16* Wt = (_Float16*)smem;   // [256][264] transposed W, f16

  int mat = blockIdx.z;                       // 0: X2@W1 -> whk1, 1: X1@W2 -> whk2
  const float* X = mat ? X1 : X2;
  const float* W = mat ? W2 : W1;
  _Float16* out = mat ? whk2 : whk1;

  int t = threadIdx.x;
  // stage W transposed into LDS: Wt[n][k] = W[k][n]; global reads coalesced per k
  for (int k = 0; k < DD; ++k) Wt[t * 264 + k] = (_Float16)W[k * DD + t];
  __syncthreads();

  int w = t >> 5;
  int lane = t & 31;
  int n0 = (w & 3) * 64;            // feature tile for this wave
  int msub = w >> 2;                // 0..1: which of the block's two m-streams
  int akk = (lane < 16) ? 0 : 8;    // A frag K sub-offset (ISA 16-bit A layout)
  int bkk = (lane < 16) ? 0 : 8;    // B frag K sub-offset

  for (int mt = 0; mt < M_ITERS; ++mt) {
    int m0 = ((blockIdx.x * M_ITERS + mt) * 2 + msub) * 16;   // 0..16368
    int b = m0 / NN;
    int ml = m0 % NN;
    int arow = m0 + (lane & 15);

    f8 acc[4];
    #pragma unroll
    for (int c = 0; c < 4; ++c) acc[c] = (f8){};

    const float* xrow = X + (size_t)arow * DD;
    for (int k0 = 0; k0 < DD; k0 += 32) {
      // A fragment: halves 0..7 = K(kk+0..7), halves 8..15 = K(kk+16..23)
      f4 x0 = *(const f4*)(xrow + k0 + akk);
      f4 x1 = *(const f4*)(xrow + k0 + akk + 4);
      f4 x2 = *(const f4*)(xrow + k0 + akk + 16);
      f4 x3 = *(const f4*)(xrow + k0 + akk + 20);
      h16 af;
      #pragma unroll
      for (int i = 0; i < 4; ++i) { af[i] = (_Float16)x0[i]; af[4 + i] = (_Float16)x1[i];
                                    af[8 + i] = (_Float16)x2[i]; af[12 + i] = (_Float16)x3[i]; }
      #pragma unroll
      for (int c = 0; c < 4; ++c) {
        const _Float16* wp = Wt + (size_t)(n0 + c * 16 + (lane & 15)) * 264 + k0 + bkk;
        h8 lo = *(const h8*)wp;
        h8 hi = *(const h8*)(wp + 16);
        h16 bf;
        #pragma unroll
        for (int i = 0; i < 8; ++i) { bf[i] = lo[i]; bf[8 + i] = hi[i]; }
        acc[c] = __builtin_amdgcn_wmma_f32_16x16x32_f16(false, af, false, bf,
                                                        (short)0, acc[c], false, false);
      }
    }
    // transposed f16 store: C element r is row M = r + (lane<16?0:8) -> contiguous
    // in the key dim of whk[b][feat][key]; one 16B store per tile per lane.
    // Regular (RT) stores on purpose: whk is re-read 128x by the attention GEMM
    // and we want it resident in L2.
    #pragma unroll
    for (int c = 0; c < 4; ++c) {
      int n = n0 + c * 16 + (lane & 15);
      _Float16* op = out + ((size_t)b * DD + n) * NN + ml + ((lane < 16) ? 0 : 8);
      h8 st;
      #pragma unroll
      for (int r = 0; r < 8; ++r) st[r] = (_Float16)acc[c][r];
      *(h8*)op = st;
    }
  }
}

// --------------------------------------------------------------------------
// K3: fused softmax + probs writeback + ctx GEMM.
// One block = one (dir, batch, 16-query tile). 256 threads (8 waves).
// LDS: pe[16][2052] unnormalized rows (128KB) + sk/mask staging (16KB).
// Phase 1a: e = lrelu(sq_i + sk_j) + mask_j, exact block row-max.
// Phase 1b: p = exp(e - m), row sums -> 1/denom.  (exp evaluated once)
// Phase 2a: probs = p/denom, nontemporal f32 stores (streaming, 128KB/block).
// Phase 2b: ctx = P @ Wh_k via v_wmma_f32_16x16x32_f16; each wave owns 32 of
//           256 output features; A built from LDS scaled by invd, B from the
//           transposed f16 whk (contiguous 16B loads, L2-resident, prefetched
//           4 k-steps ahead with global_prefetch).
// --------------------------------------------------------------------------
__global__ __launch_bounds__(256)
void attn_kernel(const float* __restrict__ sq1, const float* __restrict__ sk1,
                 const float* __restrict__ sq2, const float* __restrict__ sk2,
                 const float* __restrict__ mask1, const float* __restrict__ mask2,
                 const _Float16* __restrict__ whk1, const _Float16* __restrict__ whk2,
                 float* __restrict__ out) {
  extern __shared__ char smem[];
  float* pe    = (float*)smem;            // [16][PE_PITCH]
  float* skv   = pe + 16 * PE_PITCH;      // [2048]
  float* mkv   = skv + NN;                // [2048]
  float* wred  = mkv + NN;                // [16][8]
  float* mrow  = wred + 128;              // [16]
  float* invd  = mrow + 16;               // [16]
  float* sqrow = invd + 16;               // [16]

  int dir = blockIdx.z, b = blockIdx.y, q0 = blockIdx.x * 16;
  const float* sq = dir ? sq2 : sq1;
  const float* sk = dir ? sk2 : sk1;
  const float* mk = dir ? mask1 : mask2;      // probs1 uses mask2, probs2 uses mask1
  const _Float16* whk = dir ? whk2 : whk1;
  float* ctx   = out + (size_t)dir * BB * NN * DD;
  float* probs = out + (size_t)2 * BB * NN * DD + (size_t)dir * BB * NN * NN;

  int t = threadIdx.x;
  for (int j = t; j < NN; j += 256) {
    skv[j] = sk[b * NN + j];
    mkv[j] = mk[b * NN + j];
  }
  if (t < 16) sqrow[t] = sq[b * NN + q0 + t];
  __syncthreads();

  int w = t >> 5, lane = t & 31;

  // ---- phase 1a: scores + exact row max --------------------------------
  for (int r = 0; r < 16; ++r) {
    float sqr = sqrow[r];
    float lm = -3.0e38f;
    #pragma unroll
    for (int c = 0; c < 8; ++c) {
      int col = t + 256 * c;
      float x = sqr + skv[col];
      float e = (x > 0.f ? x : ALPHA * x) + mkv[col];
      pe[r * PE_PITCH + col] = e;
      lm = fmaxf(lm, e);
    }
    #pragma unroll
    for (int off = 16; off > 0; off >>= 1) lm = fmaxf(lm, __shfl_xor(lm, off, 32));
    if (lane == 0) wred[r * 8 + w] = lm;
  }
  __syncthreads();
  if (t < 16) {
    float m = wred[t * 8];
    #pragma unroll
    for (int i = 1; i < 8; ++i) m = fmaxf(m, wred[t * 8 + i]);
    mrow[t] = m;
  }
  __syncthreads();

  // ---- phase 1b: exp once, row sums ------------------------------------
  for (int r = 0; r < 16; ++r) {
    float m = mrow[r];
    float s = 0.f;
    #pragma unroll
    for (int c = 0; c < 8; ++c) {
      int col = t + 256 * c;
      float p = __expf(pe[r * PE_PITCH + col] - m);
      pe[r * PE_PITCH + col] = p;
      s += p;
    }
    #pragma unroll
    for (int off = 16; off > 0; off >>= 1) s += __shfl_xor(s, off, 32);
    if (lane == 0) wred[r * 8 + w] = s;
  }
  __syncthreads();
  if (t < 16) {
    float s = 0.f;
    #pragma unroll
    for (int i = 0; i < 8; ++i) s += wred[t * 8 + i];
    invd[t] = 1.0f / s;
  }
  __syncthreads();

  // ---- phase 2a: normalized probs to HBM (coalesced, nontemporal) ------
  for (int r = 0; r < 16; ++r) {
    float id = invd[r];
    float* pr = probs + ((size_t)b * NN + q0 + r) * NN;
    #pragma unroll
    for (int c = 0; c < 8; ++c) {
      int col = t + 256 * c;
      __builtin_nontemporal_store(pe[r * PE_PITCH + col] * id, pr + col);
    }
  }

  // ---- phase 2b: ctx = P @ Wh_k via WMMA (pe is read-only now) ---------
  int n0 = w * 32;                 // this wave's 32 output features
  int arow = lane & 15;
  int akk = (lane < 16) ? 0 : 8;
  int bkk = (lane < 16) ? 0 : 8;
  float id = invd[arow];
  f8 acc0 = (f8){}, acc1 = (f8){};

  const _Float16* brow0 = whk + ((size_t)b * DD + n0 + (lane & 15)) * NN + bkk;
  const _Float16* brow1 = brow0 + (size_t)16 * NN;

  for (int k0 = 0; k0 < NN; k0 += 32) {
    const float* pp = pe + arow * PE_PITCH + k0 + akk;
    h16 af;
    #pragma unroll
    for (int i = 0; i < 8; ++i) af[i] = (_Float16)(pp[i] * id);
    #pragma unroll
    for (int i = 0; i < 8; ++i) af[8 + i] = (_Float16)(pp[16 + i] * id);

    const _Float16* bp0 = brow0 + k0;
    const _Float16* bp1 = brow1 + k0;
    // pull the cachelines for k0+128 into flight (global_prefetch_b8)
    __builtin_prefetch((const void*)(bp0 + 128), 0, 3);
    __builtin_prefetch((const void*)(bp1 + 128), 0, 3);
    h8 l0 = *(const h8*)bp0, h0 = *(const h8*)(bp0 + 16);
    h8 l1 = *(const h8*)bp1, h1 = *(const h8*)(bp1 + 16);
    h16 bf0, bf1;
    #pragma unroll
    for (int i = 0; i < 8; ++i) { bf0[i] = l0[i]; bf0[8 + i] = h0[i];
                                  bf1[i] = l1[i]; bf1[8 + i] = h1[i]; }
    acc0 = __builtin_amdgcn_wmma_f32_16x16x32_f16(false, af, false, bf0, (short)0, acc0, false, false);
    acc1 = __builtin_amdgcn_wmma_f32_16x16x32_f16(false, af, false, bf1, (short)0, acc1, false, false);
  }

  // store ctx tile (16 queries x 32 features for this wave), streaming -> NT
  int mhi = (lane < 16) ? 0 : 8;
  #pragma unroll
  for (int c = 0; c < 2; ++c) {
    f8 a = c ? acc1 : acc0;
    int n = n0 + c * 16 + (lane & 15);
    #pragma unroll
    for (int r = 0; r < 8; ++r)
      __builtin_nontemporal_store(a[r], ctx + ((size_t)b * NN + q0 + mhi + r) * DD + n);
  }
}

// --------------------------------------------------------------------------
extern "C" void kernel_launch(void* const* d_in, const int* in_sizes, int n_in,
                              void* d_out, int out_size, void* d_ws, size_t ws_size,
                              hipStream_t stream) {
  const float* X1 = (const float*)d_in[0];
  const float* M1 = (const float*)d_in[1];
  const float* X2 = (const float*)d_in[2];
  const float* M2 = (const float*)d_in[3];
  const float* W1 = (const float*)d_in[4];
  const float* a1 = (const float*)d_in[5];
  const float* W2 = (const float*)d_in[6];
  const float* a2 = (const float*)d_in[7];

  // workspace layout (~17MB): two transposed f16 key matrices + score vectors
  char* ws = (char*)d_ws;
  _Float16* whk1 = (_Float16*)ws;                                // 8 MB
  _Float16* whk2 = (_Float16*)(ws + (size_t)BB * DD * NN * 2);   // 8 MB
  float* sq1 = (float*)(ws + (size_t)2 * BB * DD * NN * 2);
  float* sk1 = sq1 + BB * NN;
  float* sq2 = sk1 + BB * NN;
  float* sk2 = sq2 + BB * NN;
  float* vecs = sk2 + BB * NN;                                   // 4 x 256

  constexpr int LDS_PROJ = 256 * 264 * 2;  // 135,168 B
  constexpr int LDS_ATTN = (16 * PE_PITCH + 2 * NN + 128 + 48) * 4;  // 148,416 B
  hipFuncSetAttribute((const void*)proj_kernel,
                      hipFuncAttributeMaxDynamicSharedMemorySize, LDS_PROJ);
  hipFuncSetAttribute((const void*)attn_kernel,
                      hipFuncAttributeMaxDynamicSharedMemorySize, LDS_ATTN);

  fold_kernel<<<dim3(4), dim3(256), 0, stream>>>(W1, a1, W2, a2, vecs);
  score_kernel<<<dim3(8192), dim3(256), 0, stream>>>(X1, X2, vecs, sq1, sk1, sq2, sk2);
  proj_kernel<<<dim3(64, 1, 2), dim3(256), LDS_PROJ, stream>>>(X1, X2, W1, W2, whk1, whk2);
  attn_kernel<<<dim3(128, 8, 2), dim3(256), LDS_ATTN, stream>>>(
      sq1, sk1, sq2, sk2, M1, M2, whk1, whk2, (float*)d_out);
}
